// PRmoduleBlock_8366596293082
// MI455X (gfx1250) — matrix operand order repaired and checked by reading, not compiled, not run on hardware
//
// PRmodule block (VoxelMorph-style) for MI455X / gfx1250.
// Pipeline: repack weights -> memset halos -> warp -> corr27/assemble ->
// 3x implicit-GEMM conv via v_wmma_f32_16x16x32_f16
// (M=Cout=16, N=16 x-voxels, K=Cin*27 chunked by 32).
// Staged activations live on a 98^3 zero-padded grid so the 27-tap conv loop
// has NO branches / exec masking: every load is base + immediate offset.
#include <hip/hip_runtime.h>

typedef _Float16 h16v __attribute__((ext_vector_type(16)));
typedef _Float16 h8v  __attribute__((ext_vector_type(8)));
typedef float    f8v  __attribute__((ext_vector_type(8)));

static constexpr int   Bb  = 2, Chn = 16, Dd = 96, Hh = 96, Ww = 96;
static constexpr int   PD  = 98;                                // padded dim
static constexpr long long VOX  = (long long)Dd * Hh * Ww;      // 884736
static constexpr long long NV   = (long long)Bb * VOX;          // 1769472
static constexpr long long PVOX = (long long)Bb * PD * PD * PD; // 1882384
static constexpr float LEAK = 0.2f;

__device__ __forceinline__ float lrelu(float x) { return fmaxf(x, LEAK * x); }

// ---------------------------------------------------------------------------
// Weight repack into A-fragment layout for V_WMMA_F32_16X16X32_F16.
// A is 16(M=Cout) x 32(K=cin chunk). Lanes 0-15 (M=lane) hold halves
// K0..15; lanes 16-31 (M=lane-16) hold K16..31 (matches B contiguous order).
// afrag[t][chunk][lane][half], t = (kd*3+kh)*3+kw.
// ---------------------------------------------------------------------------
__global__ __launch_bounds__(256) void k_repack(const float* __restrict__ w,
                                                int Cout, int Cin, int nch,
                                                _Float16* __restrict__ afrag) {
  int tid = blockIdx.x * blockDim.x + threadIdx.x;
  int total = 27 * nch * 32;
  if (tid >= total) return;
  int l  = tid & 31;
  int c  = (tid >> 5) % nch;
  int t  = tid / (32 * nch);
  int hi = l >> 4;
  int M  = l & 15;
  _Float16 vals[16];
#pragma unroll
  for (int h = 0; h < 16; ++h) {
    int K   = h + 16 * hi;
    int cin = c * 32 + K;
    float v = 0.0f;
    if (M < Cout && cin < Cin)
      v = w[((long long)M * Cin + cin) * 27 + t];
    vals[h] = (_Float16)v;
  }
  *(h16v*)(afrag + ((long long)(t * nch + c) * 32 + l) * 16) = *(h16v*)vals;
}

// ---------------------------------------------------------------------------
// Kernel 1: trilinear DVF upsample (48^3 -> 96^3, align_corners) + warp.
// Writes warped[vox][16] fp16 (channels-last, unpadded grid).
// ---------------------------------------------------------------------------
__global__ __launch_bounds__(256) void k_warp(const float* __restrict__ src,
                                              const float* __restrict__ dvf,
                                              _Float16* __restrict__ warped) {
  long long v = (long long)blockIdx.x * 256 + threadIdx.x;
  if (v >= NV) return;
  int x = (int)(v % Ww);
  int y = (int)((v / Ww) % Hh);
  int z = (int)((v / (Ww * Hh)) % Dd);
  int b = (int)(v / VOX);

  // --- upsample dvf (scale (48-1)/(96-1)) ---
  const float s = 47.0f / 95.0f;
  float pz = z * s, py = y * s, px = x * s;
  int iz0 = (int)pz, iy0 = (int)py, ix0 = (int)px;
  float fz = pz - iz0, fy = py - iy0, fx = px - ix0;
  int iz1 = min(iz0 + 1, 47), iy1 = min(iy0 + 1, 47), ix1 = min(ix0 + 1, 47);
  float wz0 = 1.0f - fz, wy0 = 1.0f - fy, wx0 = 1.0f - fx;
  float flow[3];
#pragma unroll
  for (int ch = 0; ch < 3; ++ch) {
    const float* dp = dvf + (long long)(b * 3 + ch) * 48 * 48 * 48;
    auto at = [&](int zz, int yy, int xx) { return dp[(zz * 48 + yy) * 48 + xx]; };
    flow[ch] =
        wz0 * (wy0 * (wx0 * at(iz0, iy0, ix0) + fx * at(iz0, iy0, ix1)) +
               fy  * (wx0 * at(iz0, iy1, ix0) + fx * at(iz0, iy1, ix1))) +
        fz  * (wy0 * (wx0 * at(iz1, iy0, ix0) + fx * at(iz1, iy0, ix1)) +
               fy  * (wx0 * at(iz1, iy1, ix0) + fx * at(iz1, iy1, ix1)));
  }

  // --- trilinear sample of source at (z,y,x)+flow, zeros padding ---
  float sz = z + flow[0], sy = y + flow[1], sx = x + flow[2];
  float z0f = floorf(sz), y0f = floorf(sy), x0f = floorf(sx);
  int z0 = (int)z0f, y0 = (int)y0f, x0 = (int)x0f;
  float gz = sz - z0f, gy = sy - y0f, gx = sx - x0f;

  float acc[Chn];
#pragma unroll
  for (int c = 0; c < Chn; ++c) acc[c] = 0.0f;

#pragma unroll
  for (int k = 0; k < 8; ++k) {
    int zi = z0 + ((k >> 2) & 1);
    int yi = y0 + ((k >> 1) & 1);
    int xi = x0 + (k & 1);
    float wgt = (((k >> 2) & 1) ? gz : 1.0f - gz) *
                (((k >> 1) & 1) ? gy : 1.0f - gy) *
                ((k & 1) ? gx : 1.0f - gx);
    bool valid = (unsigned)zi < (unsigned)Dd && (unsigned)yi < (unsigned)Hh &&
                 (unsigned)xi < (unsigned)Ww;
    if (valid && wgt != 0.0f) {
      const float* sp = src + (long long)b * Chn * VOX +
                        (long long)zi * (Hh * Ww) + yi * Ww + xi;
#pragma unroll
      for (int c = 0; c < Chn; ++c) acc[c] += wgt * sp[(long long)c * VOX];
    }
  }
  h16v o;
#pragma unroll
  for (int c = 0; c < Chn; ++c) o[c] = (_Float16)acc[c];
  *(h16v*)(warped + v * Chn) = o;
}

// ---------------------------------------------------------------------------
// Kernel 2: 27-shift correlation + assemble feature fp16 on the PADDED grid:
// feat[b][z+1][y+1][x+1][64]: ch 0..15 = source, 16..42 = lrelu(corr27),
// 43..58 = target, 59..63 = 0. Halo voxels were zeroed by memset.
// ---------------------------------------------------------------------------
__global__ __launch_bounds__(256) void k_corr(const float* __restrict__ src,
                                              const float* __restrict__ tgt,
                                              const _Float16* __restrict__ warped,
                                              _Float16* __restrict__ feat) {
  long long v = (long long)blockIdx.x * 256 + threadIdx.x;
  if (v >= NV) return;
  int x = (int)(v % Ww);
  int y = (int)((v / Ww) % Hh);
  int z = (int)((v / (Ww * Hh)) % Dd);
  int b = (int)(v / VOX);

  long long cb = (long long)b * Chn * VOX + (long long)z * (Hh * Ww) + y * Ww + x;
  float t16[Chn], s16[Chn];
#pragma unroll
  for (int c = 0; c < Chn; ++c) {
    t16[c] = tgt[cb + (long long)c * VOX];
    s16[c] = src[cb + (long long)c * VOX];
  }

  h16v ov[4];
  _Float16* o = (_Float16*)ov;
#pragma unroll
  for (int c = 0; c < Chn; ++c) o[c] = (_Float16)s16[c];

  int idx = 16;
#pragma unroll
  for (int dz = -1; dz <= 1; ++dz)
#pragma unroll
    for (int dy = -1; dy <= 1; ++dy)
#pragma unroll
      for (int dx = -1; dx <= 1; ++dx) {
        int zz = z + dz, yy = y + dy, xx = x + dx;
        float sum = 0.0f;
        if ((unsigned)zz < (unsigned)Dd && (unsigned)yy < (unsigned)Hh &&
            (unsigned)xx < (unsigned)Ww) {
          long long nb = (long long)b * VOX + (long long)zz * (Hh * Ww) + yy * Ww + xx;
          h16v wv = *(const h16v*)(warped + nb * Chn);
#pragma unroll
          for (int c = 0; c < Chn; ++c) sum += t16[c] * (float)wv[c];
        }
        o[idx++] = (_Float16)lrelu(sum * (1.0f / Chn));
      }
#pragma unroll
  for (int c = 0; c < Chn; ++c) o[43 + c] = (_Float16)t16[c];
#pragma unroll
  for (int c = 59; c < 64; ++c) o[c] = (_Float16)0.0f;

  long long pv = (((long long)b * PD + (z + 1)) * PD + (y + 1)) * PD + (x + 1);
  h16v* fp = (h16v*)(feat + pv * 64);
#pragma unroll
  for (int i = 0; i < 4; ++i) fp[i] = ov[i];
}

// ---------------------------------------------------------------------------
// Implicit-GEMM 3x3x3 conv via WMMA on the zero-padded grid. One wave = one
// (b,z,y,16-x-tile) output tile of all 16 Cout channels. Fully unrolled taps:
// every A/B load is one base pointer + compile-time immediate offset; no
// branches, EXEC all-ones throughout.
// B layout: column N = lane%16 (voxel), lanes 0-15 hold K=0..15, lanes 16-31
// hold K=16..31 (contiguous channels -> one 32B per-lane load).
// EPI: 0 = write fp16[pvox][32] (upper 16 zero); 1 = +residual, same store;
//      2 = write 3 fp32 channels to d_out.
// ---------------------------------------------------------------------------
template <int CPAD, int NCH, int EPI>
__global__ __launch_bounds__(256) void k_conv(const _Float16* __restrict__ fin,
                                              const _Float16* __restrict__ afrag,
                                              const float* __restrict__ bias,
                                              _Float16* __restrict__ fout,
                                              const _Float16* __restrict__ resid,
                                              float* __restrict__ out) {
  int lane = threadIdx.x & 31;
  int wib  = threadIdx.x >> 5;
  long long wid = (long long)blockIdx.x * 8 + wib;

  int xt = (int)(wid % 6);
  long long r1 = wid / 6;
  int y = (int)(r1 % Hh);
  long long r2 = r1 / Hh;
  int z = (int)(r2 % Dd);
  int b = (int)(r2 / Dd);
  if (b >= Bb) return;  // wave-uniform guard

  int n  = lane & 15;
  int hi = lane >> 4;
  int x  = xt * 16 + n;

  // Base = padded voxel (b, z, y, x) == tap (tz,ty,tx) = (0,0,0); center is +1.
  const _Float16* bbase =
      fin + ((((long long)b * PD + z) * PD + y) * PD + x) * CPAD + hi * 16;
  const _Float16* abase = afrag + (long long)lane * 16;

  f8v acc = {};
#pragma unroll
  for (int tz = 0; tz < 3; ++tz)
#pragma unroll
    for (int ty = 0; ty < 3; ++ty)
#pragma unroll
      for (int tx = 0; tx < 3; ++tx) {
        const int t = (tz * 3 + ty) * 3 + tx;
#pragma unroll
        for (int c = 0; c < NCH; ++c) {
          h16v av = *(const h16v*)(abase + (t * NCH + c) * 512);
          h16v bv = *(const h16v*)(bbase +
                                   ((tz * PD * PD + ty * PD + tx) * CPAD + c * 32));
          acc = __builtin_amdgcn_wmma_f32_16x16x32_f16(false, av, false, bv,
                                                       (short)0, acc, false, false);
        }
      }

  long long pvc = (((long long)b * PD + (z + 1)) * PD + (y + 1)) * PD + (x + 1);
  if (EPI == 0 || EPI == 1) {
    h8v o;
#pragma unroll
    for (int r = 0; r < 8; ++r) {
      int m = r + 8 * hi;  // output channel
      float val = lrelu(acc[r] + bias[m]);
      if (EPI == 1) val += (float)resid[pvc * 32 + m];
      o[r] = (_Float16)val;
    }
    *(h8v*)(fout + pvc * 32 + hi * 8) = o;          // channels m..m+7
    h8v zz8 = {};
    *(h8v*)(fout + pvc * 32 + 16 + hi * 8) = zz8;   // zero pad ch 16..31
  } else {
    if (hi == 0) {
#pragma unroll
      for (int r = 0; r < 3; ++r) {
        float val = lrelu(acc[r] + bias[r]);
        out[((long long)(b * 3 + r)) * VOX + (long long)z * (Hh * Ww) + y * Ww + x] =
            val;
      }
    }
  }
}

// ---------------------------------------------------------------------------
extern "C" void kernel_launch(void* const* d_in, const int* in_sizes, int n_in,
                              void* d_out, int out_size, void* d_ws, size_t ws_size,
                              hipStream_t stream) {
  const float* src = (const float*)d_in[0];
  const float* tgt = (const float*)d_in[1];
  const float* dvf = (const float*)d_in[2];
  const float* w1  = (const float*)d_in[3];
  const float* b1  = (const float*)d_in[4];
  const float* w2  = (const float*)d_in[5];
  const float* b2  = (const float*)d_in[6];
  const float* w3  = (const float*)d_in[7];
  const float* b3  = (const float*)d_in[8];

  _Float16* ws     = (_Float16*)d_ws;
  _Float16* warped = ws;                    // NV*16 halves   ( 56.6 MB)
  _Float16* feat   = warped + NV * 16;      // PVOX*64 halves (241.0 MB)
  _Float16* f      = feat + PVOX * 64;      // PVOX*32 halves (120.5 MB)
  _Float16* f2     = f + PVOX * 32;         // PVOX*32 halves (120.5 MB)
  _Float16* a1     = f2 + PVOX * 32;        // 27*2*512 halves
  _Float16* a2     = a1 + 27 * 2 * 512;     // 27*1*512 halves
  _Float16* a3     = a2 + 27 * 1 * 512;     // 27*1*512 halves

  // Zero the padded staging buffers (feat | f | f2 are contiguous) so conv
  // halo taps read zeros with no branching.
  hipMemsetAsync(feat, 0, (size_t)(PVOX * (64 + 32 + 32)) * sizeof(_Float16),
                 stream);

  // Weight repack into WMMA A-fragment layout (zero-padded K / M).
  k_repack<<<(27 * 2 * 32 + 255) / 256, 256, 0, stream>>>(w1, 16, 59, 2, a1);
  k_repack<<<(27 * 1 * 32 + 255) / 256, 256, 0, stream>>>(w2, 16, 16, 1, a2);
  k_repack<<<(27 * 1 * 32 + 255) / 256, 256, 0, stream>>>(w3, 3, 16, 1, a3);

  int vblocks = (int)((NV + 255) / 256);  // 6912
  k_warp<<<vblocks, 256, 0, stream>>>(src, dvf, warped);
  k_corr<<<vblocks, 256, 0, stream>>>(src, tgt, warped, feat);

  long long waves = (long long)Bb * Dd * Hh * (Ww / 16);  // 110592
  int cblocks = (int)(waves / 8);                         // 13824
  k_conv<64, 2, 0><<<cblocks, 256, 0, stream>>>(feat, a1, b1, f, nullptr, nullptr);
  k_conv<32, 1, 1><<<cblocks, 256, 0, stream>>>(f, a2, b2, f2, f, nullptr);
  k_conv<32, 1, 2><<<cblocks, 256, 0, stream>>>(f2, a3, b3, nullptr, nullptr,
                                                (float*)d_out);
}
